// PCL_73263552135295
// MI455X (gfx1250) — compile-verified, matrix operand order checked by reading, not run on hardware
//
#include <hip/hip_runtime.h>
#include <hip/hip_bf16.h>

typedef __attribute__((ext_vector_type(16))) __bf16 v16bf;
typedef __attribute__((ext_vector_type(8)))  float  v8f;

#define BATCH   16384
#define DIN     2048
#define DF      128
#define NCLS    10
#define LOG2W   (-0.014499569695115089f)   // log2(0.99)
#define ONE_MW  (0.01f)                    // 1 - 0.99

// ---------------------------------------------------------------------------
// Kernel 1: W_enc [2048 x 128] f32  ->  Wt [128 x 2048] bf16 (transposed)
// ---------------------------------------------------------------------------
__global__ __launch_bounds__(256)
void PCL_convert_wenc(const float* __restrict__ W, __bf16* __restrict__ Wt) {
    int idx = blockIdx.x * 256 + threadIdx.x;      // over 2048*128
    int k = idx >> 7;
    int n = idx & 127;
    Wt[(long)n * DIN + k] = (__bf16)W[idx];
}

// ---------------------------------------------------------------------------
// Kernel 2/3: feat = l2norm(img @ W_enc) via bf16 WMMA; output = feat @ W_cls;
//             optional argmax -> pseudo labels.
// 256 threads = 8 waves; 128 rows per workgroup; wave owns 16 rows x 128 cols.
// Double-buffered LDS staging (1 barrier/iter), staging global loads issued at
// the top of the iteration, B-operand register ping-pong under the WMMA chain.
// ---------------------------------------------------------------------------
__global__ __launch_bounds__(256)
void PCL_encode(const float* __restrict__ img,
                const __bf16* __restrict__ Wt,     // [128][2048] bf16
                const float* __restrict__ Wcls,    // [128][10]
                float* __restrict__ featOut,       // [BATCH][128]
                float* __restrict__ outOut,        // [BATCH][10]
                float* __restrict__ pseudoF,       // [BATCH] or null
                int*   __restrict__ pseudoI)       // [BATCH] or null
{
    // [buf][col][32 K] bf16, padded to 40 bf16 (80B) per col so the per-lane
    // ds_load_b128 operand fetches are bank-conflict free.
    __shared__ __bf16 Bt[2][128 * 40];

    const int tid  = threadIdx.x;
    const int wave = tid >> 5;
    const int lane = tid & 31;
    const int l16  = lane & 15;
    const int half = lane >> 4;
    const int rowBase = blockIdx.x * 128 + wave * 16;
    const long rowOff = (long)(rowBase + l16) * DIN;

    // staging unit assignment: 512 x 16B units, 2 per thread
    const int u0 = tid,        col0 = u0 >> 2, sub0 = u0 & 3;
    const int u1 = tid + 256,  col1 = u1 >> 2, sub1 = u1 & 3;
    const int ldsOff0 = col0 * 40 + sub0 * 8;   // in bf16 elements
    const int ldsOff1 = col1 * 40 + sub1 * 8;
    const __bf16* gsrc0 = Wt + (long)col0 * DIN + sub0 * 8;
    const __bf16* gsrc1 = Wt + (long)col1 * DIN + sub1 * 8;

    v8f c[8];
#pragma unroll
    for (int i = 0; i < 8; ++i)
#pragma unroll
        for (int j = 0; j < 8; ++j) c[i][j] = 0.f;

    // prologue: stage chunk k0=0 into buffer 0
    *((uint4*)(&Bt[0][ldsOff0])) = *(const uint4*)(gsrc0);
    *((uint4*)(&Bt[0][ldsOff1])) = *(const uint4*)(gsrc1);
    __syncthreads();

    for (int k0 = 0; k0 < DIN; k0 += 32) {
        const int cur = (k0 >> 5) & 1;
        const int nxt = cur ^ 1;
        const bool haveNext = (k0 + 32 < DIN);

        // issue staging global loads for the NEXT chunk first (latency hidden
        // under this iteration's convert + 8 WMMAs)
        uint4 s0, s1;
        if (haveNext) {
            s0 = *(const uint4*)(gsrc0 + k0 + 32);
            s1 = *(const uint4*)(gsrc1 + k0 + 32);
            __builtin_prefetch(img + rowOff + k0 + 32 + half * 8, 0, 3);
        }

        // A operand: row = rowBase + l16; K segs [k0+half*8, +8) and +16.
        const float4* ap0 = (const float4*)(img + rowOff + k0 + half * 8);
        float4 a0 = ap0[0], a1 = ap0[1];
        const float4* ap1 = (const float4*)(img + rowOff + k0 + 16 + half * 8);
        float4 a2 = ap1[0], a3 = ap1[1];
        v16bf a;
        a[0]=(__bf16)a0.x; a[1]=(__bf16)a0.y; a[2]=(__bf16)a0.z; a[3]=(__bf16)a0.w;
        a[4]=(__bf16)a1.x; a[5]=(__bf16)a1.y; a[6]=(__bf16)a1.z; a[7]=(__bf16)a1.w;
        a[8]=(__bf16)a2.x; a[9]=(__bf16)a2.y; a[10]=(__bf16)a2.z; a[11]=(__bf16)a2.w;
        a[12]=(__bf16)a3.x; a[13]=(__bf16)a3.y; a[14]=(__bf16)a3.z; a[15]=(__bf16)a3.w;

        // B ping-pong: load nt+1 before wmma(nt) so each ds_load pair has a
        // full WMMA of latency cover.
        const __bf16* bbase = &Bt[cur][0];
        union { uint4 q[2]; v16bf v; } ub[2];
        {
            const __bf16* bp = bbase + l16 * 40 + half * 8;
            ub[0].q[0] = *(const uint4*)bp;
            ub[0].q[1] = *(const uint4*)(bp + 16);
        }
#pragma unroll
        for (int nt = 0; nt < 8; ++nt) {
            if (nt < 7) {
                const __bf16* bp = bbase + ((nt + 1) * 16 + l16) * 40 + half * 8;
                ub[(nt + 1) & 1].q[0] = *(const uint4*)bp;
                ub[(nt + 1) & 1].q[1] = *(const uint4*)(bp + 16);
            }
            c[nt] = __builtin_amdgcn_wmma_f32_16x16x32_bf16(
                false, a, false, ub[nt & 1].v, (short)0, c[nt], false, false);
        }

        // commit staged registers into the other LDS buffer
        if (haveNext) {
            *((uint4*)(&Bt[nxt][ldsOff0])) = s0;
            *((uint4*)(&Bt[nxt][ldsOff1])) = s1;
        }
        __syncthreads();
    }

    // ---- L2-normalize rows entirely in-wave ----
    float scl[8];
#pragma unroll
    for (int j = 0; j < 8; ++j) {
        float s = 0.f;
#pragma unroll
        for (int nt = 0; nt < 8; ++nt) { float v = c[nt][j]; s += v * v; }
#pragma unroll
        for (int m = 1; m < 16; m <<= 1) s += __shfl_xor(s, m, 32);
        scl[j] = 1.0f / fmaxf(sqrtf(s), 1e-12f);
    }
#pragma unroll
    for (int nt = 0; nt < 8; ++nt)
#pragma unroll
        for (int j = 0; j < 8; ++j) c[nt][j] *= scl[j];

    // ---- store normalized feat ----
#pragma unroll
    for (int nt = 0; nt < 8; ++nt) {
        int n = nt * 16 + l16;
#pragma unroll
        for (int j = 0; j < 8; ++j) {
            int m = rowBase + j + half * 8;
            featOut[(long)m * DF + n] = c[nt][j];
        }
    }

    // ---- output = feat @ W_cls, plus argmax ----
    float best[8]; int bidx[8];
#pragma unroll
    for (int j = 0; j < 8; ++j) { best[j] = -3.402823466e+38f; bidx[j] = 0; }

    for (int cls = 0; cls < NCLS; ++cls) {
        float wv[8];
#pragma unroll
        for (int nt = 0; nt < 8; ++nt)
            wv[nt] = Wcls[(nt * 16 + l16) * NCLS + cls];
#pragma unroll
        for (int j = 0; j < 8; ++j) {
            float p = 0.f;
#pragma unroll
            for (int nt = 0; nt < 8; ++nt) p += c[nt][j] * wv[nt];
#pragma unroll
            for (int m = 1; m < 16; m <<= 1) p += __shfl_xor(p, m, 32);
            if (l16 == 0)
                outOut[(long)(rowBase + j + half * 8) * NCLS + cls] = p;
            if (p > best[j]) { best[j] = p; bidx[j] = cls; }
        }
    }
    if (pseudoF != nullptr && l16 == 0) {
#pragma unroll
        for (int j = 0; j < 8; ++j) {
            int m = rowBase + j + half * 8;
            pseudoF[m] = (float)bidx[j];
            pseudoI[m] = bidx[j];
        }
    }
}

// ---------------------------------------------------------------------------
// Kernel 4: in-class ranks -> EMA coefficients alpha_i = 0.01 * w^(k_c - r_i)
// Single workgroup, 256 threads, 64 samples/thread; both label arrays.
// ---------------------------------------------------------------------------
__global__ __launch_bounds__(256)
void PCL_rank(const int* __restrict__ labP, const int* __restrict__ labY,
              float* __restrict__ alphaP, float* __restrict__ alphaY,
              int* __restrict__ counts /* [20] */)
{
    __shared__ int cnt[256][NCLS];
    __shared__ int tot[NCLS];
    const int t = threadIdx.x;
    const int base = t * 64;

    for (int pass = 0; pass < 2; ++pass) {
        const int* lab = pass ? labY : labP;
        float* alpha   = pass ? alphaY : alphaP;
        __syncthreads();

        int local[NCLS];
#pragma unroll
        for (int cc = 0; cc < NCLS; ++cc) local[cc] = 0;
        for (int i = 0; i < 64; ++i) {
            int l = lab[base + i];
#pragma unroll
            for (int cc = 0; cc < NCLS; ++cc) local[cc] += (l == cc) ? 1 : 0;
        }
#pragma unroll
        for (int cc = 0; cc < NCLS; ++cc) cnt[t][cc] = local[cc];
        __syncthreads();

        if (t < NCLS) {
            int run = 0;
            for (int tt = 0; tt < 256; ++tt) {
                int v = cnt[tt][t];
                cnt[tt][t] = run;
                run += v;
            }
            tot[t] = run;
            counts[pass * NCLS + t] = run;
        }
        __syncthreads();

        int run[NCLS], totv[NCLS];
#pragma unroll
        for (int cc = 0; cc < NCLS; ++cc) { run[cc] = cnt[t][cc]; totv[cc] = tot[cc]; }
        for (int i = 0; i < 64; ++i) {
            int l = lab[base + i];
            float e = 0.f;
#pragma unroll
            for (int cc = 0; cc < NCLS; ++cc) {
                if (l == cc) { run[cc] += 1; e = (float)(totv[cc] - run[cc]); }
            }
            alpha[base + i] = ONE_MW * exp2f(e * LOG2W);
        }
    }
}

// ---------------------------------------------------------------------------
// Kernel 5a: per-block weighted class partial sums via LDS float atomics.
// grid 64 x 256 threads; block handles 256 samples (2 samples in flight).
// ---------------------------------------------------------------------------
__global__ __launch_bounds__(256)
void PCL_proto_partial(const float* __restrict__ feat1,
                       const int* __restrict__ labP, const int* __restrict__ labY,
                       const float* __restrict__ alphaP, const float* __restrict__ alphaY,
                       float* __restrict__ partP,    // [64][10][128]
                       float* __restrict__ partY)    // [64][10][128]
{
    __shared__ float pp[NCLS * DF];
    __shared__ float py[NCLS * DF];
    const int t = threadIdx.x;
    for (int i = t; i < NCLS * DF; i += 256) { pp[i] = 0.f; py[i] = 0.f; }
    __syncthreads();

    const int d   = t & 127;
    const int sub = t >> 7;
    const int base = blockIdx.x * 256;
    for (int rr = 0; rr < 128; ++rr) {
        int i = base + rr * 2 + sub;
        float f = feat1[(long)i * DF + d];
        int   lp = labP[i];  float ap = alphaP[i];
        int   ly = labY[i];  float ay = alphaY[i];
        atomicAdd(&pp[lp * DF + d], ap * f);
        atomicAdd(&py[ly * DF + d], ay * f);
    }
    __syncthreads();
    for (int i = t; i < NCLS * DF; i += 256) {
        partP[blockIdx.x * (NCLS * DF) + i] = pp[i];
        partY[blockIdx.x * (NCLS * DF) + i] = py[i];
    }
}

// ---------------------------------------------------------------------------
// Kernel 5b: finalize protos: p = w^k * p0 + sum(partials); L2-normalize row.
// grid 20 (bank*10 + class), 128 threads (feature dim).
// ---------------------------------------------------------------------------
__global__ __launch_bounds__(128)
void PCL_proto_final(const float* __restrict__ partP, const float* __restrict__ partY,
                     const float* __restrict__ p0P, const float* __restrict__ p0Y,
                     const int* __restrict__ counts,
                     float* __restrict__ protoT, float* __restrict__ protoC)
{
    const int b = blockIdx.x;
    const int bank = b / NCLS;
    const int c = b % NCLS;
    const float* part = bank ? partY : partP;
    const float* p0   = bank ? p0Y   : p0P;
    float* dst        = bank ? protoC : protoT;
    const int d = threadIdx.x;

    int k = counts[bank * NCLS + c];
    float wk = exp2f((float)k * LOG2W);
    float acc = wk * p0[c * DF + d];
    for (int blk = 0; blk < 64; ++blk)
        acc += part[blk * (NCLS * DF) + c * DF + d];

    __shared__ float red[DF];
    red[d] = acc * acc;
    __syncthreads();
    for (int s = 64; s > 0; s >>= 1) {
        if (d < s) red[d] += red[d + s];
        __syncthreads();
    }
    float scale = 1.0f / fmaxf(sqrtf(red[0]), 1e-12f);
    dst[c * DF + d] = acc * scale;
}

// ---------------------------------------------------------------------------
// Kernel 6: logits_prot = feat2@protoT^T, logits_prot2 = feat1@protoT^T,
//           logits_com = feat2@protoC^T.  grid 64 x 256, thread per row.
// ---------------------------------------------------------------------------
__global__ __launch_bounds__(256)
void PCL_logits(const float* __restrict__ feat1, const float* __restrict__ feat2,
                const float* __restrict__ protoT, const float* __restrict__ protoC,
                float* __restrict__ logitsProt, float* __restrict__ logitsProt2,
                float* __restrict__ logitsCom)
{
    __shared__ float pt[NCLS * DF];
    __shared__ float pc[NCLS * DF];
    const int t = threadIdx.x;
    for (int i = t; i < NCLS * DF; i += 256) { pt[i] = protoT[i]; pc[i] = protoC[i]; }
    __syncthreads();

    long row = (long)blockIdx.x * 256 + t;
    const float* f1 = feat1 + row * DF;
    const float* f2 = feat2 + row * DF;
    float a1[NCLS], a2[NCLS], a3[NCLS];
#pragma unroll
    for (int cc = 0; cc < NCLS; ++cc) { a1[cc] = 0.f; a2[cc] = 0.f; a3[cc] = 0.f; }
    for (int d = 0; d < DF; ++d) {
        float x1 = f1[d], x2 = f2[d];
#pragma unroll
        for (int cc = 0; cc < NCLS; ++cc) {
            float p = pt[cc * DF + d];
            a1[cc] += x2 * p;
            a2[cc] += x1 * p;
            a3[cc] += x2 * pc[cc * DF + d];
        }
    }
#pragma unroll
    for (int cc = 0; cc < NCLS; ++cc) {
        logitsProt [row * NCLS + cc] = a1[cc];
        logitsProt2[row * NCLS + cc] = a2[cc];
        logitsCom  [row * NCLS + cc] = a3[cc];
    }
}

// ---------------------------------------------------------------------------
// Host launch
// ---------------------------------------------------------------------------
extern "C" void kernel_launch(void* const* d_in, const int* in_sizes, int n_in,
                              void* d_out, int out_size, void* d_ws, size_t ws_size,
                              hipStream_t stream) {
    const float* img1   = (const float*)d_in[0];
    const float* img2   = (const float*)d_in[1];
    const int*   y      = (const int*)d_in[2];
    const float* W_enc  = (const float*)d_in[3];
    const float* W_cls  = (const float*)d_in[4];
    const float* protoT0 = (const float*)d_in[5];
    const float* protoC0 = (const float*)d_in[6];

    float* out = (float*)d_out;
    // output tuple offsets (floats)
    float* o_output      = out;                       // [16384,10]
    float* o_output2     = out + 163840;              // [16384,10]
    float* o_logitsProt  = out + 327680;              // [16384,10]
    float* o_logitsProt2 = out + 491520;              // [16384,10]
    float* o_pseudo      = out + 655360;              // [16384]
    float* o_logitsCom   = out + 671744;              // [16384,10]
    float* o_feat1       = out + 835584;              // [16384,128]

    // workspace layout (bytes)
    char* ws = (char*)d_ws;
    __bf16* Wt      = (__bf16*)(ws);                        // 512 KB
    float*  feat2   = (float*)(ws + 524288);                // 8 MB
    int*    pseudoI = (int*)(ws + 8912896);                 // 64 KB
    float*  alphaP  = (float*)(ws + 8978432);               // 64 KB
    float*  alphaY  = (float*)(ws + 9043968);               // 64 KB
    int*    counts  = (int*)(ws + 9109504);                 // 128 B (20 used)
    float*  partP   = (float*)(ws + 9109632);               // 320 KB
    float*  partY   = (float*)(ws + 9437312);               // 320 KB
    float*  protoT  = (float*)(ws + 9764992);               // 5 KB
    float*  protoC  = (float*)(ws + 9770112);               // 5 KB

    // 1) convert + transpose W_enc -> bf16
    PCL_convert_wenc<<<(DIN * DF) / 256, 256, 0, stream>>>(W_enc, Wt);

    // 2) encode img1 (feat1 -> d_out, output, pseudo labels)
    PCL_encode<<<BATCH / 128, 256, 0, stream>>>(
        img1, Wt, W_cls, o_feat1, o_output, o_pseudo, pseudoI);

    // 3) encode img2 (feat2 -> ws, output2)
    PCL_encode<<<BATCH / 128, 256, 0, stream>>>(
        img2, Wt, W_cls, feat2, o_output2, nullptr, nullptr);

    // 4) ranks -> EMA coefficients
    PCL_rank<<<1, 256, 0, stream>>>(pseudoI, y, alphaP, alphaY, counts);

    // 5) proto partial sums + finalize (EMA closed form) + L2-normalize
    PCL_proto_partial<<<64, 256, 0, stream>>>(
        o_feat1, pseudoI, y, alphaP, alphaY, partP, partY);
    PCL_proto_final<<<20, 128, 0, stream>>>(
        partP, partY, protoT0, protoC0, counts, protoT, protoC);

    // 6) prototype logits
    PCL_logits<<<BATCH / 256, 256, 0, stream>>>(
        o_feat1, feat2, protoT, protoC, o_logitsProt, o_logitsProt2, o_logitsCom);
}
